// ByteLatentTransformer_82927228551673
// MI455X (gfx1250) — compile-verified, matrix operand order checked by reading, not run on hardware
//
#include <hip/hip_runtime.h>
#include <hip/hip_bf16.h>

#define B_    4
#define S_    2048
#define D_    512
#define P_    512
#define NH_   8
#define DH_   64
#define MINP_ 4
#define MAXP_ 32
#define EWIN_ 8
#define ETH_  2.0f

typedef __bf16 v16bf __attribute__((ext_vector_type(16)));
typedef float  v8f   __attribute__((ext_vector_type(8)));

// ---------- helpers ----------
__device__ __forceinline__ unsigned short f2bf(float f) {
  unsigned int u = __builtin_bit_cast(unsigned int, f);
  u += 0x7FFFu + ((u >> 16) & 1u);                  // round-to-nearest-even
  return (unsigned short)(u >> 16);
}

// K index of the v-th packed bf16 pair for a lane in half {0,1}
// (ISA 7.12.2: 16-bit A 16x32 -> VGPR0..3: K=0..7(+8*half), VGPR4..7: K=16..23(+8*half))
__device__ __forceinline__ int kpair(int v, int half) {
  return ((v < 4) ? (2 * v) : (16 + 2 * (v - 4))) + 8 * half;
}

// A/B fragment from a row-major matrix row (K contiguous) -> 2x b128 loads
__device__ __forceinline__ v16bf frag_row(const unsigned short* base, int stride,
                                          int row, int k0, int half) {
  union { v16bf v; unsigned int u[8]; } f;
  const unsigned short* r = base + (long)row * stride + k0;
#pragma unroll
  for (int i = 0; i < 8; ++i)
    f.u[i] = *(const unsigned int*)(r + kpair(i, half));
  return f.v;
}

// fragment from an LDS row (K contiguous) -> 2x ds_load_b128
__device__ __forceinline__ v16bf frag_lds_row(const unsigned short* lr, int half) {
  union { v16bf v; unsigned int u[8]; } f;
#pragma unroll
  for (int i = 0; i < 8; ++i)
    f.u[i] = *(const unsigned int*)(lr + kpair(i, half));
  return f.v;
}

// fragment where K runs down LDS rows (column gather, for P@V)
__device__ __forceinline__ v16bf frag_lds_col(const unsigned short* basecol,
                                              int ldstride, int half) {
  union { v16bf v; unsigned short s[16]; } f;
#pragma unroll
  for (int i = 0; i < 8; ++i) {
    int k = kpair(i, half);
    f.s[2 * i]     = basecol[(k)     * ldstride];
    f.s[2 * i + 1] = basecol[(k + 1) * ldstride];
  }
  return f.v;
}

__device__ __forceinline__ v8f wmma_bf16(v16bf a, v16bf b, v8f c) {
  return __builtin_amdgcn_wmma_f32_16x16x32_bf16(false, a, false, b, (short)0, c,
                                                 false, false);
}

// CDNA5 async DMA: global -> LDS, 16B per lane, tracked by ASYNCcnt
__device__ __forceinline__ void async_load_b128(unsigned lds_off, const void* g) {
  asm volatile("global_load_async_to_lds_b128 %0, %1, off"
               :: "v"(lds_off), "v"(g) : "memory");
}
__device__ __forceinline__ void wait_async0() {
  asm volatile("s_wait_asynccnt 0x0" ::: "memory");
}

// ---------- 1. sliding-window entropy ----------
__global__ void entropy_kernel(const int* __restrict__ seq, float* __restrict__ ent) {
  int t = blockIdx.x * blockDim.x + threadIdx.x;
  if (t >= B_ * S_) return;
  int b = t / S_, s = t % S_;
  int sc = min(s, S_ - EWIN_);          // replicate-last padding
  int w[EWIN_];
#pragma unroll
  for (int i = 0; i < EWIN_; ++i) w[i] = seq[b * S_ + sc + i];
  float e = 0.0f;
#pragma unroll
  for (int i = 0; i < EWIN_; ++i) {
    int cnt = 0;
#pragma unroll
    for (int j = 0; j < EWIN_; ++j) cnt += (w[j] == w[i]);
    e += log2f((float)cnt * (1.0f / EWIN_) + 1e-10f);
  }
  ent[t] = -e * (1.0f / EWIN_);
}

// ---------- 2. sequential segmentation scan (one wave per batch) ----------
__global__ void segment_kernel(const float* __restrict__ ent,
                               int* __restrict__ starts, int* __restrict__ ends) {
  int b = threadIdx.x >> 5;
  int lane = threadIdx.x & 31;
  if (b >= B_) return;
  const float* e = ent + b * S_;
  int start = 0;
  for (int p = 0; p < P_; ++p) {
    int end0 = min(start + MAXP_, S_);
    float val = __builtin_inff();
    int idx = start + MINP_ + lane;
    if (lane < (MAXP_ - MINP_) && idx < end0) {
      int ci = min(max(idx, 0), S_ - 1);
      val = e[ci];
    }
    int mi = lane;
    for (int off = 16; off >= 1; off >>= 1) {      // argmin, first-index tiebreak
      float ov = __shfl_xor(val, off, 32);
      int   oi = __shfl_xor(mi, off, 32);
      if (ov < val || (ov == val && oi < mi)) { val = ov; mi = oi; }
    }
    bool use_b = (start + MAXP_ < S_) && (val < ETH_);
    int end = use_b ? (start + MINP_ + mi + 1) : end0;
    if (start >= S_) end = start;
    if (lane == 0) { starts[b * P_ + p] = start; ends[b * P_ + p] = end; }
    start = end;
  }
}

// ---------- 3. patches / lengths / positions ----------
__global__ void patches_kernel(const int* __restrict__ seq,
                               const int* __restrict__ starts,
                               const int* __restrict__ ends,
                               float* __restrict__ out) {
  int t = blockIdx.x * blockDim.x + threadIdx.x;
  if (t >= B_ * P_) return;
  const int OFF_LEN = B_ * P_ * MAXP_;
  const int OFF_POS = OFF_LEN + B_ * P_;
  int b = t / P_;
  int st = starts[t], en = ends[t], len = en - st;
  out[OFF_LEN + t] = (float)len;
  out[OFF_POS + 2 * t]     = (float)((len == 0) ? 0 : st);
  out[OFF_POS + 2 * t + 1] = (float)((len == 0) ? 0 : en);
#pragma unroll
  for (int j = 0; j < MAXP_; ++j) {
    int idx = min(max(st + j, 0), S_ - 1);
    out[t * MAXP_ + j] = (float)((j < len) ? seq[b * S_ + idx] : 0);
  }
}

// ---------- 4a. layernorm -> bf16 (one wave per row) ----------
__global__ void ln_kernel(const float* __restrict__ x, const float* __restrict__ g,
                          const float* __restrict__ be, unsigned short* __restrict__ y) {
  int row = (blockIdx.x * blockDim.x + threadIdx.x) >> 5;
  int lane = threadIdx.x & 31;
  if (row >= B_ * P_) return;
  const float* r = x + (long)row * D_;
  float s = 0.0f, ss = 0.0f;
  for (int i = lane; i < D_; i += 32) { float v = r[i]; s += v; ss += v * v; }
  for (int off = 16; off >= 1; off >>= 1) {
    s += __shfl_xor(s, off, 32);
    ss += __shfl_xor(ss, off, 32);
  }
  float mu = s * (1.0f / D_);
  float var = ss * (1.0f / D_) - mu * mu;
  float rs = rsqrtf(var + 1e-5f);
  for (int i = lane; i < D_; i += 32)
    y[(long)row * D_ + i] = f2bf((r[i] - mu) * rs * g[i] + be[i]);
}

// ---------- 4b. fp32 -> bf16 ----------
__global__ void cvt_bf16_kernel(const float* __restrict__ x,
                                unsigned short* __restrict__ y, int n) {
  int i = blockIdx.x * blockDim.x + threadIdx.x;
  if (i < n) y[i] = f2bf(x[i]);
}

// ---------- 5. cooperative bf16 WMMA GEMM with async-LDS weight staging ----------
// C[m][n] = sum_k A[m][k]*W[n][k] + bias[n]; K compile-time, loop fully unrolled.
// Block: 8 waves sharing columns n0..n0+63; wave w owns rows mstrip+32w..+31.
template <int K>
__global__ __launch_bounds__(256) void gemm32_bf16_kernel(
    const unsigned short* __restrict__ A, const unsigned short* __restrict__ W,
    const float* __restrict__ bias, unsigned short* __restrict__ C,
    int M, int N, int nblk) {
  __shared__ unsigned short wtile[2][64][32];          // 2 x 4KB slabs
  const int tid = threadIdx.x;
  const int lane = tid & 31;
  const int wv = tid >> 5;
  const int half = lane >> 4;
  const int ln = lane & 15;
  const int mstrip = (blockIdx.x / nblk) * 256;
  const int n0 = (blockIdx.x % nblk) * 64;
  const int row0 = mstrip + wv * 32 + ln;
  const int row1 = row0 + 16;
  const int wrow = tid >> 2;                           // 0..63: weight row
  const int wq = tid & 3;                              // 16B quarter of 64B row

  // stage 0 DMA issue
  async_load_b128((unsigned)(size_t)&wtile[0][wrow][wq * 8],
                  W + (size_t)(n0 + wrow) * K + wq * 8);

  v16bf a0 = frag_row(A, K, row0, 0, half);
  v16bf a1 = frag_row(A, K, row1, 0, half);

  v8f acc[2][4];
#pragma unroll
  for (int r = 0; r < 2; ++r)
#pragma unroll
    for (int t = 0; t < 4; ++t) acc[r][t] = (v8f){0, 0, 0, 0, 0, 0, 0, 0};

#pragma unroll
  for (int kk = 0; kk < K / 32; ++kk) {
    const int k0 = kk * 32;
    const int cur = kk & 1;
    wait_async0();                 // my slab writes have landed
    __syncthreads();               // everyone's slab visible; prev reads retired
    if (kk + 1 < K / 32)           // DMA next slab while computing on this one
      async_load_b128((unsigned)(size_t)&wtile[cur ^ 1][wrow][wq * 8],
                      W + (size_t)(n0 + wrow) * K + (k0 + 32) + wq * 8);
    v16bf a0n = a0, a1n = a1;      // next A fragments (SSA after full unroll)
    if (kk + 1 < K / 32) {
      a0n = frag_row(A, K, row0, k0 + 32, half);
      a1n = frag_row(A, K, row1, k0 + 32, half);
    }
    v16bf bf[4];                   // issue all 8 ds_load_b128 first
#pragma unroll
    for (int t = 0; t < 4; ++t)
      bf[t] = frag_lds_row(&wtile[cur][16 * t + ln][0], half);
#pragma unroll
    for (int t = 0; t < 4; ++t) {
      acc[0][t] = wmma_bf16(a0, bf[t], acc[0][t]);     // B-frag reused across
      acc[1][t] = wmma_bf16(a1, bf[t], acc[1][t]);     // both row halves
    }
    a0 = a0n; a1 = a1n;
  }
#pragma unroll
  for (int r = 0; r < 2; ++r)
#pragma unroll
    for (int t = 0; t < 4; ++t) {
      int col = n0 + 16 * t + ln;
      float bv = bias[col];
#pragma unroll
      for (int v = 0; v < 8; ++v)
        C[(long)(mstrip + wv * 32 + r * 16 + v + 8 * half) * N + col] =
            f2bf(acc[r][t][v] + bv);
    }
}

// ---------- 6. flash-style masked cross-attention ----------
// Block = 8 waves covering 8 consecutive 16-patch tiles of ONE (b,h): all waves
// share the same K/V 32x64 chunk, staged in LDS by async DMA (double buffered).
__global__ __launch_bounds__(256) void attn_kernel(
    const unsigned short* __restrict__ Qb, const unsigned short* __restrict__ Kb,
    const unsigned short* __restrict__ Vb, const int* __restrict__ starts,
    const int* __restrict__ ends, unsigned short* __restrict__ Ob) {
  __shared__ float sP[8][16 * 32];                      // per-wave prob tile, 16KB
  __shared__ unsigned short ktile[2][32][64];           // 8KB
  __shared__ unsigned short vtile[2][32][64];           // 8KB
  const int tid = threadIdx.x;
  const int lane = tid & 31;
  const int lwid = tid >> 5;
  const int half = lane >> 4;
  const int ln = lane & 15;
  const int blk = blockIdx.x;                           // 128 blocks
  const int mt = (blk & 3) * 8 + lwid;                  // 32 patch tiles
  const int h = (blk >> 2) & 7;
  const int b = blk >> 5;
  const int m0 = mt * 16;
  const int srow = tid >> 3;                            // 0..31: staging row
  const int soct = tid & 7;                             // 16B octant of 128B row

  // stage 0 DMA (K and V chunks, 4KB each)
  {
    const size_t gr = (size_t)(b * S_ + srow) * D_ + h * DH_ + soct * 8;
    async_load_b128((unsigned)(size_t)&ktile[0][srow][soct * 8], Kb + gr);
    async_load_b128((unsigned)(size_t)&vtile[0][srow][soct * 8], Vb + gr);
  }

  v16bf qa[2];
  qa[0] = frag_row(Qb, D_, b * P_ + m0 + ln, h * DH_ + 0, half);
  qa[1] = frag_row(Qb, D_, b * P_ + m0 + ln, h * DH_ + 32, half);

  int rs[8], re[8];
#pragma unroll
  for (int v = 0; v < 8; ++v) {
    int r = b * P_ + m0 + v + 8 * half;
    rs[v] = starts[r];
    re[v] = ends[r];
  }
  float rmax[8], rsum[8];
#pragma unroll
  for (int v = 0; v < 8; ++v) { rmax[v] = -3.0e38f; rsum[v] = 0.0f; }
  v8f o[4];
#pragma unroll
  for (int t = 0; t < 4; ++t) o[t] = (v8f){0, 0, 0, 0, 0, 0, 0, 0};
  const float scale = 0.125f;                            // 1/sqrt(64)

  for (int s0 = 0; s0 < S_; s0 += 32) {
    const int cur = (s0 >> 5) & 1;
    wait_async0();
    __syncthreads();
    if (s0 + 32 < S_) {
      const size_t gr = (size_t)(b * S_ + s0 + 32 + srow) * D_ + h * DH_ + soct * 8;
      async_load_b128((unsigned)(size_t)&ktile[cur ^ 1][srow][soct * 8], Kb + gr);
      async_load_b128((unsigned)(size_t)&vtile[cur ^ 1][srow][soct * 8], Vb + gr);
    }
    // scores: B-frags straight from LDS (2x ds_load_b128 each)
    v16bf kf[2][2];
#pragma unroll
    for (int ch = 0; ch < 2; ++ch)
#pragma unroll
      for (int ks = 0; ks < 2; ++ks)
        kf[ch][ks] = frag_lds_row(&ktile[cur][16 * ch + ln][32 * ks], half);
    v8f sc[2];
    sc[0] = (v8f){0, 0, 0, 0, 0, 0, 0, 0};
    sc[1] = (v8f){0, 0, 0, 0, 0, 0, 0, 0};
#pragma unroll
    for (int ch = 0; ch < 2; ++ch)
#pragma unroll
      for (int ks = 0; ks < 2; ++ks)
        sc[ch] = wmma_bf16(qa[ks], kf[ch][ks], sc[ch]);

    float corr[8];
#pragma unroll
    for (int v = 0; v < 8; ++v) {
      int c0 = s0 + ln, c1 = s0 + 16 + ln;
      float x0 = (c0 >= rs[v] && c0 < re[v]) ? sc[0][v] * scale : -1.0e9f;
      float x1 = (c1 >= rs[v] && c1 < re[v]) ? sc[1][v] * scale : -1.0e9f;
      float mx = fmaxf(x0, x1);                          // row max within half
      for (int off = 1; off < 16; off <<= 1) mx = fmaxf(mx, __shfl_xor(mx, off, 32));
      float nmax = fmaxf(rmax[v], mx);
      corr[v] = __expf(rmax[v] - nmax);
      rmax[v] = nmax;
      float e0 = __expf(x0 - nmax), e1 = __expf(x1 - nmax);
      sP[lwid][(v + 8 * half) * 32 + ln] = e0;
      sP[lwid][(v + 8 * half) * 32 + 16 + ln] = e1;
      float p2 = e0 + e1;
      for (int off = 1; off < 16; off <<= 1) p2 += __shfl_xor(p2, off, 32);
      rsum[v] = rsum[v] * corr[v] + p2;
    }
#pragma unroll
    for (int t = 0; t < 4; ++t)
#pragma unroll
      for (int v = 0; v < 8; ++v) o[t][v] *= corr[v];
    // C-layout -> A-layout relayout through LDS (in-order DS pipe, same wave)
    union { v16bf v; unsigned short s[16]; } pf;
#pragma unroll
    for (int i = 0; i < 8; ++i) {
      int k = kpair(i, half);
      pf.s[2 * i]     = f2bf(sP[lwid][ln * 32 + k]);
      pf.s[2 * i + 1] = f2bf(sP[lwid][ln * 32 + k + 1]);
    }
    v16bf vf[4];
#pragma unroll
    for (int t = 0; t < 4; ++t)
      vf[t] = frag_lds_col(&vtile[cur][0][16 * t + ln], 64, half);
#pragma unroll
    for (int t = 0; t < 4; ++t) o[t] = wmma_bf16(pf.v, vf[t], o[t]);
  }
#pragma unroll
  for (int t = 0; t < 4; ++t)
#pragma unroll
    for (int v = 0; v < 8; ++v) {
      float den = rsum[v];
      float val = (den > 0.0f) ? o[t][v] / den : 0.0f;
      int r = b * P_ + m0 + v + 8 * half;
      Ob[(long)r * D_ + h * DH_ + 16 * t + ln] = f2bf(val);
    }
}

// ---------- 7. out-projection + bias + any_valid mask + residual (f32 out) ----------
template <int K>
__global__ __launch_bounds__(256) void outproj_kernel(
    const unsigned short* __restrict__ A, const unsigned short* __restrict__ W,
    const float* __restrict__ bias, const float* __restrict__ resid,
    const int* __restrict__ starts, const int* __restrict__ ends,
    float* __restrict__ out, int M, int N, int ntiles) {
  const int lane = threadIdx.x & 31;
  const int half = lane >> 4;
  const int ln = lane & 15;
  const int wid = (blockIdx.x * blockDim.x + threadIdx.x) >> 5;
  const int m0 = (wid / ntiles) * 16;
  const int n0 = (wid % ntiles) * 64;
  if (m0 >= M) return;
  v8f acc[4];
#pragma unroll
  for (int t = 0; t < 4; ++t) acc[t] = (v8f){0, 0, 0, 0, 0, 0, 0, 0};
  v16bf a = frag_row(A, K, m0 + ln, 0, half);
#pragma unroll
  for (int kk = 0; kk < K / 32; ++kk) {
    const int k0 = kk * 32;
    v16bf an = a;
    if (kk + 1 < K / 32) an = frag_row(A, K, m0 + ln, k0 + 32, half);
    v16bf bw[4];
#pragma unroll
    for (int t = 0; t < 4; ++t)
      bw[t] = frag_row(W, K, n0 + 16 * t + ln, k0, half);
#pragma unroll
    for (int t = 0; t < 4; ++t) acc[t] = wmma_bf16(a, bw[t], acc[t]);
    a = an;
  }
#pragma unroll
  for (int t = 0; t < 4; ++t) {
    int col = n0 + 16 * t + ln;
    float bv = bias[col];
#pragma unroll
    for (int v = 0; v < 8; ++v) {
      int r = m0 + v + 8 * half;
      bool valid = ends[r] > starts[r];                 // r == b*P_ + p
      float ov = valid ? acc[t][v] + bv : 0.0f;
      out[(long)r * N + col] = resid[(long)r * N + col] + ov;
    }
  }
}

// ---------- host ----------
extern "C" void kernel_launch(void* const* d_in, const int* in_sizes, int n_in,
                              void* d_out, int out_size, void* d_ws, size_t ws_size,
                              hipStream_t stream) {
  const int*   seq   = (const int*)d_in[0];
  const float* pemb  = (const float*)d_in[1];
  const float* bhid  = (const float*)d_in[2];
  const float* ln_g  = (const float*)d_in[3];
  const float* ln_b  = (const float*)d_in[4];
  const float* w_in  = (const float*)d_in[5];
  const float* b_in  = (const float*)d_in[6];
  const float* w_out = (const float*)d_in[7];
  const float* b_out = (const float*)d_in[8];
  float* out = (float*)d_out;

  // workspace layout (256B aligned)
  constexpr size_t OFF_ENT  = 0;                                   // B*S f32
  constexpr size_t OFF_ST   = OFF_ENT + (size_t)B_ * S_ * 4;       // B*P i32
  constexpr size_t OFF_EN   = OFF_ST + (size_t)B_ * P_ * 4;
  constexpr size_t OFF_QIN  = OFF_EN + (size_t)B_ * P_ * 4;        // B*P*D bf16
  constexpr size_t OFF_BH   = OFF_QIN + (size_t)B_ * P_ * D_ * 2;  // B*S*D bf16
  constexpr size_t OFF_WIN  = OFF_BH + (size_t)B_ * S_ * D_ * 2;   // 3D*D bf16
  constexpr size_t OFF_WOUT = OFF_WIN + (size_t)3 * D_ * D_ * 2;   // D*D bf16
  constexpr size_t OFF_Q    = OFF_WOUT + (size_t)D_ * D_ * 2;      // B*P*D bf16
  constexpr size_t OFF_K    = OFF_Q + (size_t)B_ * P_ * D_ * 2;    // B*S*D bf16
  constexpr size_t OFF_V    = OFF_K + (size_t)B_ * S_ * D_ * 2;
  constexpr size_t OFF_O    = OFF_V + (size_t)B_ * S_ * D_ * 2;    // B*P*D bf16

  char* ws = (char*)d_ws;
  float* ent = (float*)(ws + OFF_ENT);
  int* starts = (int*)(ws + OFF_ST);
  int* ends = (int*)(ws + OFF_EN);
  unsigned short* q_in = (unsigned short*)(ws + OFF_QIN);
  unsigned short* bh_bf = (unsigned short*)(ws + OFF_BH);
  unsigned short* w_in_bf = (unsigned short*)(ws + OFF_WIN);
  unsigned short* w_out_bf = (unsigned short*)(ws + OFF_WOUT);
  unsigned short* Qb = (unsigned short*)(ws + OFF_Q);
  unsigned short* Kb = (unsigned short*)(ws + OFF_K);
  unsigned short* Vb = (unsigned short*)(ws + OFF_V);
  unsigned short* Ob = (unsigned short*)(ws + OFF_O);

  // 1-3: entropy, segmentation, patch outputs
  entropy_kernel<<<(B_ * S_ + 255) / 256, 256, 0, stream>>>(seq, ent);
  segment_kernel<<<1, 128, 0, stream>>>(ent, starts, ends);
  patches_kernel<<<(B_ * P_ + 255) / 256, 256, 0, stream>>>(seq, starts, ends, out);

  // 4: layernorm + bf16 conversions
  ln_kernel<<<(B_ * P_ * 32 + 255) / 256, 256, 0, stream>>>(pemb, ln_g, ln_b, q_in);
  {
    int n = B_ * S_ * D_;
    cvt_bf16_kernel<<<(n + 255) / 256, 256, 0, stream>>>(bhid, bh_bf, n);
    n = 3 * D_ * D_;
    cvt_bf16_kernel<<<(n + 255) / 256, 256, 0, stream>>>(w_in, w_in_bf, n);
    n = D_ * D_;
    cvt_bf16_kernel<<<(n + 255) / 256, 256, 0, stream>>>(w_out, w_out_bf, n);
  }

  // 5: Q/K/V projections — block covers 256 rows x 64 cols, async weight slabs
  const int nblk = D_ / 64;  // 8
  {
    int grid = (B_ * P_ / 256) * nblk;    // 64 blocks
    gemm32_bf16_kernel<D_><<<grid, 256, 0, stream>>>(
        q_in, w_in_bf, b_in, Qb, B_ * P_, D_, nblk);
    grid = (B_ * S_ / 256) * nblk;        // 256 blocks
    gemm32_bf16_kernel<D_><<<grid, 256, 0, stream>>>(
        bh_bf, w_in_bf + (size_t)D_ * D_, b_in + D_, Kb, B_ * S_, D_, nblk);
    gemm32_bf16_kernel<D_><<<grid, 256, 0, stream>>>(
        bh_bf, w_in_bf + (size_t)2 * D_ * D_, b_in + 2 * D_, Vb, B_ * S_, D_, nblk);
  }

  // 6: attention — 128 blocks x 8 waves; block = one (b,h), 8 patch tiles
  attn_kernel<<<128, 256, 0, stream>>>(Qb, Kb, Vb, starts, ends, Ob);

  // 7: out-projection + residual into d_out tail
  const int OFF_OUT = B_ * P_ * MAXP_ + B_ * P_ + B_ * P_ * 2;  // 71680
  outproj_kernel<D_><<<1024 * 32 / 256, 256, 0, stream>>>(
      Ob, w_out_bf, b_out, pemb, starts, ends, out + OFF_OUT, B_ * P_, D_, 8);
}
